// BayesRes_76605036691473
// MI455X (gfx1250) — compile-verified, matrix operand order checked by reading, not run on hardware
//
#include <hip/hip_runtime.h>
#include <hip/hip_bf16.h>

#define B_SZ 8192
#define F_SZ 32
#define H_SZ 128
#define NB_SZ 3
#define LRELU_SLOPE 0.01f
#define BN_EPS_C 1e-5f
#define LOG_PRIOR_C (-2.3025850929940457f)
#define INV_PRIOR_VAR2 50.0f /* 1 / (2*0.1^2) */

typedef __attribute__((ext_vector_type(16))) __bf16 v16bf;
typedef __attribute__((ext_vector_type(16))) unsigned short v16us;
typedef __attribute__((ext_vector_type(8))) float v8f;

union Frag { v16us u; v16bf bf; };

// native f32 -> bf16 (RNE); should lower to v_cvt_pk_bf16_f32 when paired
__device__ __forceinline__ unsigned short f2bf(float x) {
  return __builtin_bit_cast(unsigned short, (__bf16)x);
}
__device__ __forceinline__ float lrelu(float x) { return x >= 0.f ? x : LRELU_SLOPE * x; }

// ---------------------------------------------------------------- utilities
__global__ void zero_f32(float* p, size_t n) {
  size_t i = (size_t)blockIdx.x * blockDim.x + threadIdx.x;
  size_t stride = (size_t)gridDim.x * blockDim.x;
  for (; i < n; i += stride) p[i] = 0.f;
}

__global__ void copy_kl(const float* src, float* dst) {
  if (blockIdx.x == 0 && threadIdx.x == 0) dst[0] = src[0];
}

// ---------------------------------------------------- sample weights + KL
__global__ void sample_kl_kernel(const float* __restrict__ mu, const float* __restrict__ ls,
                                 const float* __restrict__ eps, float* __restrict__ w,
                                 float* __restrict__ kl_acc, float scale, int n) {
  __shared__ float red[256];
  float acc = 0.f;
  for (int i = blockIdx.x * blockDim.x + threadIdx.x; i < n; i += gridDim.x * blockDim.x) {
    float m = mu[i], l = ls[i], e = eps[i];
    float s = expf(l);
    w[i] = m + s * e;
    acc += (LOG_PRIOR_C - l) + (s * s + m * m) * INV_PRIOR_VAR2 - 0.5f;
  }
  red[threadIdx.x] = acc;
  __syncthreads();
  for (int off = 128; off > 0; off >>= 1) {
    if ((int)threadIdx.x < off) red[threadIdx.x] += red[threadIdx.x + off];
    __syncthreads();
  }
  if (threadIdx.x == 0) atomicAdd(kl_acc, red[0] * scale);
}

// --------------------------------------- input layer h0 = f*win+bin + stats
__global__ void input_layer_kernel(const float* __restrict__ fin,   // [B][F]
                                   const float* __restrict__ win,   // [F][H]
                                   const float* __restrict__ binp,  // [F][H]
                                   float* __restrict__ h0,          // [F][B][H]
                                   float* __restrict__ s_sum, float* __restrict__ s_sq) {
  const int f = blockIdx.x;
  const int c = threadIdx.x & 127;
  const int rh = threadIdx.x >> 7;  // 0..1
  const int r0 = blockIdx.y * 256;
  const float wv = win[f * H_SZ + c];
  const float bv = binp[f * H_SZ + c];
  float s1 = 0.f, s2 = 0.f;
  for (int r = rh; r < 256; r += 2) {
    int b = r0 + r;
    float x = fin[(size_t)b * F_SZ + f] * wv + bv;
    h0[((size_t)f * B_SZ + b) * H_SZ + c] = x;
    s1 += x; s2 += x * x;
  }
  __shared__ float red1[256], red2[256];
  red1[threadIdx.x] = s1; red2[threadIdx.x] = s2;
  __syncthreads();
  if (threadIdx.x < 128) {
    atomicAdd(&s_sum[f * H_SZ + c], red1[threadIdx.x] + red1[threadIdx.x + 128]);
    atomicAdd(&s_sq[f * H_SZ + c], red2[threadIdx.x] + red2[threadIdx.x + 128]);
  }
}

// --------------------------------------------- elementwise BN (+res) +lrelu
__global__ void bn_lrelu_kernel(const float* __restrict__ x, float* __restrict__ y,
                                const float* __restrict__ s_sum, const float* __restrict__ s_sq,
                                const float* __restrict__ g, const float* __restrict__ b,
                                float invB, size_t n) {
  size_t i = (size_t)blockIdx.x * blockDim.x + threadIdx.x;
  size_t stride = (size_t)gridDim.x * blockDim.x;
  for (; i < n; i += stride) {
    int c = (int)(i & (H_SZ - 1));
    int f = (int)(i >> 20);  // B*H = 1<<20
    float m = s_sum[f * H_SZ + c] * invB;
    float v = s_sq[f * H_SZ + c] * invB - m * m;
    float sc = g[f * H_SZ + c] * rsqrtf(v + BN_EPS_C);
    float sh = b[f * H_SZ + c] - m * sc;
    y[i] = lrelu(x[i] * sc + sh);
  }
}

__global__ void residual_kernel(float* __restrict__ h, const float* __restrict__ y2,
                                const float* __restrict__ s_sum, const float* __restrict__ s_sq,
                                const float* __restrict__ g, const float* __restrict__ b,
                                float invB, size_t n) {
  size_t i = (size_t)blockIdx.x * blockDim.x + threadIdx.x;
  size_t stride = (size_t)gridDim.x * blockDim.x;
  for (; i < n; i += stride) {
    int c = (int)(i & (H_SZ - 1));
    int f = (int)(i >> 20);
    float m = s_sum[f * H_SZ + c] * invB;
    float v = s_sq[f * H_SZ + c] * invB - m * m;
    float sc = g[f * H_SZ + c] * rsqrtf(v + BN_EPS_C);
    float sh = b[f * H_SZ + c] - m * sc;
    h[i] = lrelu(h[i] + (y2[i] * sc + sh));
  }
}

// ---------------------------------------------------------- WMMA GEMM core
// Y[f][b][k] = sum_h A'(X[f][b][h]) * W[f][h][k] + bias[f][k]
// A' optionally applies BN(+lrelu) using (in_sum,in_sq,bn_g,bn_b).
// Per-column sum / sumsq of Y accumulated into (out_sum,out_sq) via atomics.
// Block: 256 threads (8 waves). Wave (wm,wn): 16 rows x 64 cols -> 4 WMMA accs.
// Block tile: 64 rows x 128 cols; K loop = 4 x 32.
__global__ __launch_bounds__(256) void gemm_bn_kernel(
    const float* __restrict__ X,      // [F][B][H]
    const float* __restrict__ Wt,     // [F][H][H]  (indexed [h][k])
    const float* __restrict__ biasp,  // [F][H]
    float* __restrict__ Y,            // [F][B][H]
    const float* __restrict__ in_sum, const float* __restrict__ in_sq,
    const float* __restrict__ bn_g, const float* __restrict__ bn_b,
    float* __restrict__ out_sum, float* __restrict__ out_sq,
    int applyBnLrelu) {
  const int f = blockIdx.x;
  const int tid = threadIdx.x;
  const int lane = tid & 31;
  const int wave = tid >> 5;
  const int wm = wave & 3;  // M sub-tile (16 rows each)
  const int wn = wave >> 2; // N half (64 cols each)

  __shared__ alignas(32) unsigned short Wlds[4 * 8 * 32 * 16];  // 32KB, fragment order
  __shared__ float scale_s[H_SZ], shift_s[H_SZ], bias_s[H_SZ];

  // BN scale/shift + bias -> LDS
  for (int c = tid; c < H_SZ; c += 256) {
    float sc = 1.f, sh = 0.f;
    if (applyBnLrelu) {
      float m = in_sum[f * H_SZ + c] * (1.f / (float)B_SZ);
      float v = in_sq[f * H_SZ + c] * (1.f / (float)B_SZ) - m * m;
      sc = bn_g[f * H_SZ + c] * rsqrtf(v + BN_EPS_C);
      sh = bn_b[f * H_SZ + c] - m * sc;
    }
    scale_s[c] = sc; shift_s[c] = sh;
    bias_s[c] = biasp[f * H_SZ + c];
  }

  // W (128x128 fp32) -> bf16 LDS in B-fragment order:
  // slot = ((kt*8+nt)*32+lane)*16+e ; lane<16: n=nt*16+lane, K: e<8 -> kt*32+e, else kt*32+8+e
  //                                  lane>=16: n=nt*16+lane-16, K: e<8 -> kt*32+8+e, else kt*32+16+e
  const float* Wf = Wt + (size_t)f * H_SZ * H_SZ;
  for (int s = tid; s < 16384; s += 256) {
    int e = s & 15;
    int ln = (s >> 4) & 31;
    int nt = (s >> 9) & 7;
    int kt = s >> 12;
    int n = nt * 16 + (ln & 15);
    int koff = (ln < 16) ? ((e < 8) ? e : (8 + e)) : ((e < 8) ? (8 + e) : (16 + e));
    Wlds[s] = f2bf(Wf[(kt * 32 + koff) * H_SZ + n]);
  }
  __syncthreads();

  const int nTiles = B_SZ / 64;
  for (int bt = blockIdx.y; bt < nTiles; bt += gridDim.y) {
    const int rowBase = bt * 64 + wm * 16;
    const int rowM = rowBase + (lane & 15);
    const int kHalf = (lane < 16) ? 0 : 8;
    const float* xrow = X + ((size_t)f * B_SZ + rowM) * H_SZ;

    v8f acc[4] = {};
#pragma unroll
    for (int kt = 0; kt < 4; ++kt) {
      const int c0 = kt * 32 + kHalf;
      Frag a;
#pragma unroll
      for (int i = 0; i < 8; ++i) {
        float x0 = xrow[c0 + i];
        float x1 = xrow[c0 + 16 + i];
        if (applyBnLrelu) {
          x0 = lrelu(x0 * scale_s[c0 + i] + shift_s[c0 + i]);
          x1 = lrelu(x1 * scale_s[c0 + 16 + i] + shift_s[c0 + 16 + i]);
        }
        a.bf[i] = (__bf16)x0;
        a.bf[8 + i] = (__bf16)x1;
      }
#pragma unroll
      for (int j = 0; j < 4; ++j) {
        const int nt = wn * 4 + j;
        Frag bfr;
        bfr.u = *reinterpret_cast<const v16us*>(&Wlds[((kt * 8 + nt) * 32 + lane) * 16]);
        acc[j] = __builtin_amdgcn_wmma_f32_16x16x32_bf16(
            false, a.bf, false, bfr.bf, (short)0, acc[j], false, false);
      }
    }

    // epilogue: bias add, store fp32, per-column stats via atomics
    const int mBase = rowBase + ((lane < 16) ? 0 : 8);
#pragma unroll
    for (int j = 0; j < 4; ++j) {
      const int n = wn * 64 + j * 16 + (lane & 15);
      const float bv = bias_s[n];
      float* yrow = Y + ((size_t)f * B_SZ + mBase) * H_SZ + n;
      float s1 = 0.f, s2 = 0.f;
#pragma unroll
      for (int r = 0; r < 8; ++r) {
        float val = acc[j][r] + bv;
        yrow[(size_t)r * H_SZ] = val;
        s1 += val; s2 += val * val;
      }
      atomicAdd(&out_sum[f * H_SZ + n], s1);
      atomicAdd(&out_sq[f * H_SZ + n], s2);
    }
  }
}

// ---------------------------------------------- final contraction to out[B]
__global__ void output_kernel(const float* __restrict__ h, const float* __restrict__ wout,
                              const float* __restrict__ bout, const float* __restrict__ bias0,
                              float* __restrict__ out) {
  const int b = blockIdx.x;
  float acc = 0.f;
  for (int idx = threadIdx.x; idx < F_SZ * H_SZ; idx += 256) {
    int f = idx >> 7, c = idx & 127;
    acc += h[((size_t)f * B_SZ + b) * H_SZ + c] * wout[idx];
  }
  __shared__ float red[256];
  red[threadIdx.x] = acc;
  __syncthreads();
  for (int off = 128; off > 0; off >>= 1) {
    if ((int)threadIdx.x < off) red[threadIdx.x] += red[threadIdx.x + off];
    __syncthreads();
  }
  if (threadIdx.x == 0) {
    float s = red[0];
    for (int f = 0; f < F_SZ; ++f) s += bout[f];
    out[b] = s + bias0[0];
  }
}

// ---------------------------------------------------------------- launcher
extern "C" void kernel_launch(void* const* d_in, const int* in_sizes, int n_in,
                              void* d_out, int out_size, void* d_ws, size_t ws_size,
                              hipStream_t stream) {
  (void)in_sizes; (void)n_in; (void)out_size; (void)ws_size;
  const float* f_in = (const float*)d_in[0];
  const float* bn0_g = (const float*)d_in[25];
  const float* bn0_b = (const float*)d_in[26];
  const float* bna_g = (const float*)d_in[27];
  const float* bna_b = (const float*)d_in[28];
  const float* bnb_g = (const float*)d_in[29];
  const float* bnb_b = (const float*)d_in[30];
  const float* bias0 = (const float*)d_in[31];
  float* out = (float*)d_out;

  // ---- workspace carve-out (64-float aligned slices)
  float* ws = (float*)d_ws;
  size_t off = 0;
  auto alloc = [&](size_t n) { float* p = ws + off; off += (n + 63) & ~(size_t)63; return p; };
  float* kl_acc = alloc(1);
  float* s0_sum = alloc(F_SZ * H_SZ); float* s0_sq = alloc(F_SZ * H_SZ);
  float* sA_sum = alloc(F_SZ * H_SZ); float* sA_sq = alloc(F_SZ * H_SZ);
  float* sB_sum = alloc(F_SZ * H_SZ); float* sB_sq = alloc(F_SZ * H_SZ);
  float* w_win = alloc(F_SZ * H_SZ);  float* w_bin = alloc(F_SZ * H_SZ);
  float* w_wout = alloc(F_SZ * H_SZ); float* w_bout = alloc(F_SZ);
  float* w_ba = alloc((size_t)NB_SZ * F_SZ * H_SZ);
  float* w_bb = alloc((size_t)NB_SZ * F_SZ * H_SZ);
  float* w_Wa = alloc((size_t)NB_SZ * F_SZ * H_SZ * H_SZ);
  float* w_Wb = alloc((size_t)NB_SZ * F_SZ * H_SZ * H_SZ);
  float* hbuf = alloc((size_t)F_SZ * B_SZ * H_SZ);
  float* T1 = alloc((size_t)F_SZ * B_SZ * H_SZ);
  float* T2 = alloc((size_t)F_SZ * B_SZ * H_SZ);

  const size_t nAct = (size_t)F_SZ * B_SZ * H_SZ;
  const float invB = 1.f / (float)B_SZ;

  // zero KL + all stats
  zero_f32<<<64, 256, 0, stream>>>(kl_acc, (size_t)(w_win - kl_acc));

  // sample weights + KL (mean over given axes then sum -> scale factors)
  auto samp = [&](int i0, float* w, float scale, int n) {
    int blocks = (n + 255) / 256; if (blocks > 1024) blocks = 1024;
    sample_kl_kernel<<<blocks, 256, 0, stream>>>(
        (const float*)d_in[i0], (const float*)d_in[i0 + 1], (const float*)d_in[i0 + 2],
        w, kl_acc, scale, n);
  };
  samp(1, w_win, 1.f / H_SZ, F_SZ * H_SZ);
  samp(4, w_bin, 1.f / H_SZ, F_SZ * H_SZ);
  samp(7, w_Wa, 1.f / (H_SZ * H_SZ), NB_SZ * F_SZ * H_SZ * H_SZ);
  samp(10, w_ba, 1.f / H_SZ, NB_SZ * F_SZ * H_SZ);
  samp(13, w_Wb, 1.f / (H_SZ * H_SZ), NB_SZ * F_SZ * H_SZ * H_SZ);
  samp(16, w_bb, 1.f / H_SZ, NB_SZ * F_SZ * H_SZ);
  samp(19, w_wout, 1.f / H_SZ, F_SZ * H_SZ);
  samp(22, w_bout, 1.f, F_SZ);

  // input layer (h0 -> T1) + stats, then BN+lrelu -> hbuf
  input_layer_kernel<<<dim3(F_SZ, B_SZ / 256), 256, 0, stream>>>(
      f_in, w_win, w_bin, T1, s0_sum, s0_sq);
  bn_lrelu_kernel<<<2048, 256, 0, stream>>>(T1, hbuf, s0_sum, s0_sq, bn0_g, bn0_b, invB, nAct);

  // residual blocks
  for (int i = 0; i < NB_SZ; ++i) {
    zero_f32<<<32, 256, 0, stream>>>(sA_sum, (size_t)(w_win - sA_sum));
    gemm_bn_kernel<<<dim3(F_SZ, 32), 256, 0, stream>>>(
        hbuf, w_Wa + (size_t)i * F_SZ * H_SZ * H_SZ, w_ba + (size_t)i * F_SZ * H_SZ, T1,
        nullptr, nullptr, nullptr, nullptr, sA_sum, sA_sq, 0);
    gemm_bn_kernel<<<dim3(F_SZ, 32), 256, 0, stream>>>(
        T1, w_Wb + (size_t)i * F_SZ * H_SZ * H_SZ, w_bb + (size_t)i * F_SZ * H_SZ, T2,
        sA_sum, sA_sq, bna_g + (size_t)i * F_SZ * H_SZ, bna_b + (size_t)i * F_SZ * H_SZ,
        sB_sum, sB_sq, 1);
    residual_kernel<<<2048, 256, 0, stream>>>(
        hbuf, T2, sB_sum, sB_sq, bnb_g + (size_t)i * F_SZ * H_SZ,
        bnb_b + (size_t)i * F_SZ * H_SZ, invB, nAct);
  }

  // final contraction + KL scalar
  output_kernel<<<B_SZ, 256, 0, stream>>>(hbuf, w_wout, w_bout, bias0, out);
  copy_kl<<<1, 32, 0, stream>>>(kl_acc, out + B_SZ);
}
